// QJLKeyCache_61151744360780
// MI455X (gfx1250) — matrix operand order repaired and checked by reading, not compiled, not run on hardware
//
#include <hip/hip_runtime.h>
#include <stdint.h>

// Problem constants (from reference)
#define B_ROWS 2048
#define D_DIM  128
#define M_DIM  256
#define NKEYS  32768
#define TILE_B 64                  // rows per workgroup in score GEMM
#define TILE_N 128                 // key columns per workgroup
#define NCHUNK (NKEYS / TILE_N)    // 256 column chunks per row
#define EST_CONST 0.0048957583f    // sqrt(pi/2)/256

typedef __attribute__((ext_vector_type(16))) _Float16 v16h;
typedef __attribute__((ext_vector_type(8)))  _Float16 v8h;
typedef __attribute__((ext_vector_type(8)))  float    v8f;
typedef __attribute__((ext_vector_type(4)))  float    v4f;

// ---------------------------------------------------------------------------
// Kernel 1: convert +/-1 f32 keys -> f16, folding EST_CONST * key_norms[n].
// 8 elements per thread (M=256 divisible by 8 -> all 8 share one row/norm).
// ---------------------------------------------------------------------------
__global__ __launch_bounds__(256) void qjl_keyconv(const float* __restrict__ keys,
                                                   const float* __restrict__ norms,
                                                   _Float16* __restrict__ kf16) {
  const int idx  = blockIdx.x * 256 + threadIdx.x;
  const int base = idx * 8;               // < 8,388,608
  const int n    = base >> 8;             // / M_DIM
  const float s  = EST_CONST * norms[n];
  v4f x0 = ((const v4f*)(keys + base))[0];
  v4f x1 = ((const v4f*)(keys + base))[1];
  v8h h;
#pragma unroll
  for (int i = 0; i < 4; ++i) h[i]     = (_Float16)(x0[i] * s);
#pragma unroll
  for (int i = 0; i < 4; ++i) h[4 + i] = (_Float16)(x1[i] * s);
  *(v8h*)(kf16 + base) = h;
}

// ---------------------------------------------------------------------------
// Kernel 2: P = query @ sketch^T  -> f16 [B_ROWS, M_DIM], via WMMA.
// One wave per 16x16 output tile. K = D_DIM = 128 -> 4 steps of 32.
// A frag layout (16-bit 16x32): lane<16: row=lane, K {k0..k0+15} (contiguous);
//                               lane>=16: row=lane-16, K {+16..+31}.
// B frag (32x16): lane holds one output-column's contiguous 16 K values.
// ---------------------------------------------------------------------------
__global__ __launch_bounds__(256) void qjl_project(const float* __restrict__ query,
                                                   const float* __restrict__ sketch,
                                                   _Float16* __restrict__ pf16) {
  const int gwid = (blockIdx.x * 256 + threadIdx.x) >> 5;
  const int lane = threadIdx.x & 31;
  const int mt   = gwid & (M_DIM / 16 - 1);   // 16 column tiles
  const int bt   = gwid >> 4;                 // 128 row tiles
  const int ar   = lane & 15;
  const int akb  = (lane >> 4) * 16;
  const int bn   = lane & 15;
  const int bkb  = (lane >> 4) * 16;

  v8f acc = {};
#pragma unroll
  for (int k0 = 0; k0 < D_DIM; k0 += 32) {
    const float* qa = query + (bt * 16 + ar) * D_DIM + k0 + akb;
    v4f q0 = ((const v4f*)qa)[0];
    v4f q1 = ((const v4f*)qa)[1];
    v4f q2 = ((const v4f*)(qa + 8))[0];
    v4f q3 = ((const v4f*)(qa + 8))[1];
    v16h a;
#pragma unroll
    for (int i = 0; i < 4; ++i) {
      a[i]      = (_Float16)q0[i];
      a[4 + i]  = (_Float16)q1[i];
      a[8 + i]  = (_Float16)q2[i];
      a[12 + i] = (_Float16)q3[i];
    }
    const float* sp = sketch + (mt * 16 + bn) * D_DIM + k0 + bkb;
    v4f s0 = ((const v4f*)sp)[0];
    v4f s1 = ((const v4f*)sp)[1];
    v4f s2 = ((const v4f*)sp)[2];
    v4f s3 = ((const v4f*)sp)[3];
    v16h b;
#pragma unroll
    for (int i = 0; i < 4; ++i) {
      b[i]      = (_Float16)s0[i];
      b[4 + i]  = (_Float16)s1[i];
      b[8 + i]  = (_Float16)s2[i];
      b[12 + i] = (_Float16)s3[i];
    }
    acc = __builtin_amdgcn_wmma_f32_16x16x32_f16(false, a, false, b,
                                                 (short)0, acc, false, false);
  }
  // C layout: VGPR j -> row j (lanes 0-15) / row j+8 (lanes 16-31), col = lane&15
  const int cl = lane & 15;
  const int hi = lane >> 4;
#pragma unroll
  for (int j = 0; j < 8; ++j)
    pf16[(bt * 16 + j + 8 * hi) * M_DIM + mt * 16 + cl] = (_Float16)acc[j];
}

// ---------------------------------------------------------------------------
// Kernel 3: scores = P_f16 @ Keys_f16^T (norms pre-folded). WG tile 64x128,
// 8 waves, each wave 32x32 (2x2 WMMA tiles). A tile is staged into 32KB LDS
// with CDNA5 async-to-LDS copies (ASYNCcnt tracked, no VGPR data movement),
// then consumed via ds_load_b128. The same LDS is reused afterwards for the
// f32 C tile to compute per-row (max, sumexp) softmax partials.
// ---------------------------------------------------------------------------
__global__ __launch_bounds__(256) void qjl_score(const _Float16* __restrict__ pf16,
                                                 const _Float16* __restrict__ kf16,
                                                 float* __restrict__ out,
                                                 float2* __restrict__ partials) {
  __shared__ __align__(16) char smem[TILE_B * M_DIM * sizeof(_Float16)]; // 32 KB
  _Float16* As = (_Float16*)smem;            // [64][256] halves during GEMM
  float*    Cs = (float*)smem;               // [64][128] floats after GEMM

  const int tid    = threadIdx.x;
  const int chunk  = blockIdx.x & (NCHUNK - 1);
  const int rowblk = blockIdx.x >> 8;        // / NCHUNK
  const int row0   = rowblk * TILE_B;
  const int col0   = chunk * TILE_N;

  // Async A-tile stage: 64x256 halves = 32 KB. Each thread moves 8 x 16B
  // directly from global to LDS (GLOBAL_LOAD_ASYNC_TO_LDS_B128, GV mode).
  // LDS byte address = low 32 bits of the flat shared pointer (aperture rule).
  {
    const uint32_t lds0 = (uint32_t)(uintptr_t)As + (uint32_t)(tid * 16);
    const char*    g0   = (const char*)(pf16 + row0 * M_DIM) + tid * 16;
#pragma unroll
    for (int i = 0; i < 8; ++i) {
      uint32_t    la = lds0 + i * 4096;
      const char* ga = g0 + i * 4096;
      asm volatile("global_load_async_to_lds_b128 %0, %1, off"
                   :
                   : "v"(la), "v"(ga)
                   : "memory");
    }
    asm volatile("s_wait_asynccnt 0x0" ::: "memory");
  }
  __syncthreads();

  const int lane = tid & 31;
  const int wid  = tid >> 5;
  const int wrow = wid >> 2;                 // 0..1  -> 32-row slab
  const int wcol = wid & 3;                  // 0..3  -> 32-col slab
  const int ar   = lane & 15;
  const int akb  = (lane >> 4) * 16;
  const int bn   = lane & 15;
  const int bkb  = (lane >> 4) * 16;

  v8f acc00 = {}, acc01 = {}, acc10 = {}, acc11 = {};
  const _Float16* bbase = kf16 + (size_t)(col0 + wcol * 32) * M_DIM;

  for (int k0 = 0; k0 < M_DIM; k0 += 32) {
    v16h a0, a1, b0, b1;
    const _Float16* ap0 = As + (wrow * 32 + ar) * M_DIM + k0 + akb;
    const _Float16* ap1 = ap0 + 16 * M_DIM;
    *((v8h*)&a0)     = *(const v8h*)ap0;        // ds_load_b128
    *((v8h*)&a0 + 1) = *(const v8h*)(ap0 + 8);
    *((v8h*)&a1)     = *(const v8h*)ap1;
    *((v8h*)&a1 + 1) = *(const v8h*)(ap1 + 8);
    const _Float16* bp0 = bbase + (size_t)bn * M_DIM + k0 + bkb;
    b0 = *(const v16h*)bp0;                     // L2-resident f16 keys
    b1 = *(const v16h*)(bp0 + 16 * M_DIM);
    if (k0 + 32 < M_DIM) __builtin_prefetch(bp0 + 32, 0, 0);  // global_prefetch
    acc00 = __builtin_amdgcn_wmma_f32_16x16x32_f16(false, a0, false, b0, (short)0, acc00, false, false);
    acc01 = __builtin_amdgcn_wmma_f32_16x16x32_f16(false, a0, false, b1, (short)0, acc01, false, false);
    acc10 = __builtin_amdgcn_wmma_f32_16x16x32_f16(false, a1, false, b0, (short)0, acc10, false, false);
    acc11 = __builtin_amdgcn_wmma_f32_16x16x32_f16(false, a1, false, b1, (short)0, acc11, false, false);
  }
  __syncthreads();   // done reading As; reuse LDS as C tile

  const int cl = lane & 15;
  const int hi = lane >> 4;
#pragma unroll
  for (int j = 0; j < 8; ++j) {
    const int r0  = wrow * 32 + j + 8 * hi;
    const int r1  = r0 + 16;
    const int c0l = wcol * 32 + cl;
    const int c1l = c0l + 16;
    Cs[r0 * TILE_N + c0l] = acc00[j];
    Cs[r0 * TILE_N + c1l] = acc01[j];
    Cs[r1 * TILE_N + c0l] = acc10[j];
    Cs[r1 * TILE_N + c1l] = acc11[j];
    out[(size_t)(row0 + r0) * NKEYS + col0 + c0l] = acc00[j];
    out[(size_t)(row0 + r0) * NKEYS + col0 + c1l] = acc01[j];
    out[(size_t)(row0 + r1) * NKEYS + col0 + c0l] = acc10[j];
    out[(size_t)(row0 + r1) * NKEYS + col0 + c1l] = acc11[j];
  }
  __syncthreads();

  // Per-row softmax partials over this 128-column chunk.
  if (tid < TILE_B) {
    const float* row = Cs + tid * TILE_N;
    float mx = row[0];
#pragma unroll 4
    for (int i = 1; i < TILE_N; ++i) mx = fmaxf(mx, row[i]);
    float se = 0.0f;
#pragma unroll 4
    for (int i = 0; i < TILE_N; ++i) se += __expf(row[i] - mx);
    partials[(size_t)(row0 + tid) * NCHUNK + chunk] = make_float2(mx, se);
  }
}

// ---------------------------------------------------------------------------
// Kernel 4: reduce 256 (max, sumexp) partials per row -> (gmax, 1/gsum).
// ---------------------------------------------------------------------------
__global__ __launch_bounds__(256) void qjl_rowreduce(const float2* __restrict__ partials,
                                                     float2* __restrict__ rowstats) {
  __shared__ float smax[256];
  __shared__ float ssum[256];
  const int b   = blockIdx.x;
  const int tid = threadIdx.x;
  float2 v = partials[(size_t)b * NCHUNK + tid];   // NCHUNK == 256 == blockDim
  smax[tid] = v.x;
  ssum[tid] = v.y;
  for (int off = 128; off > 0; off >>= 1) {
    __syncthreads();
    if (tid < off) {
      const float m1 = smax[tid], m2 = smax[tid + off];
      const float s1 = ssum[tid], s2 = ssum[tid + off];
      const float m  = fmaxf(m1, m2);
      smax[tid] = m;
      ssum[tid] = s1 * __expf(m1 - m) + s2 * __expf(m2 - m);
    }
  }
  if (tid == 0) rowstats[b] = make_float2(smax[0], 1.0f / ssum[0]);
}

// ---------------------------------------------------------------------------
// Kernel 5: out = exp(s - gmax) * invsum, float4-vectorized.
// ---------------------------------------------------------------------------
__global__ __launch_bounds__(256) void qjl_softmax(float* __restrict__ out,
                                                   const float2* __restrict__ rowstats) {
  const int i4 = blockIdx.x * 256 + threadIdx.x;   // index in float4 units
  const int b  = i4 >> 13;                         // NKEYS/4 = 8192 per row
  const float2 st = rowstats[b];
  v4f x = ((v4f*)out)[i4];
#pragma unroll
  for (int i = 0; i < 4; ++i) x[i] = __expf(x[i] - st.x) * st.y;
  ((v4f*)out)[i4] = x;
}

// ---------------------------------------------------------------------------
// Launch. Workspace layout (needs ~21.1 MB):
//   [0,      1 MiB)  P_f16   [2048 x 256] f16
//   [1,     17 MiB)  K_f16   [32768 x 256] f16 (norms folded)
//   [17,    21 MiB)  partials[2048 x 256] float2
//   [21 MiB, +16KiB) rowstats[2048] float2
// ---------------------------------------------------------------------------
extern "C" void kernel_launch(void* const* d_in, const int* in_sizes, int n_in,
                              void* d_out, int out_size, void* d_ws, size_t ws_size,
                              hipStream_t stream) {
  const float* query  = (const float*)d_in[0];
  const float* keys   = (const float*)d_in[1];
  const float* norms  = (const float*)d_in[2];
  const float* sketch = (const float*)d_in[3];
  float* out = (float*)d_out;
  char* ws = (char*)d_ws;

  _Float16* pf16     = (_Float16*)ws;
  _Float16* kf16     = (_Float16*)(ws + (size_t)(1u << 20));
  float2*   partials = (float2*)(ws + (size_t)(17u << 20));
  float2*   rowstats = (float2*)(ws + (size_t)(21u << 20));

  qjl_keyconv<<<(NKEYS * M_DIM) / (8 * 256), 256, 0, stream>>>(keys, norms, kf16);
  qjl_project<<<((B_ROWS / 16) * (M_DIM / 16)) / 8, 256, 0, stream>>>(query, sketch, pf16);
  qjl_score<<<(B_ROWS / TILE_B) * NCHUNK, 256, 0, stream>>>(pf16, kf16, out, partials);
  qjl_rowreduce<<<B_ROWS, 256, 0, stream>>>(partials, rowstats);
  qjl_softmax<<<(B_ROWS * (NKEYS / 4)) / 256, 256, 0, stream>>>(out, rowstats);
}
